// Stage_26319559590247
// MI455X (gfx1250) — matrix operand order repaired and checked by reading, not compiled, not run on hardware
//
#include <hip/hip_runtime.h>
#include <hip/hip_bf16.h>
#include <math.h>

typedef __attribute__((ext_vector_type(16))) _Float16 v16h;
typedef __attribute__((ext_vector_type(8)))  float    v8f;
typedef __attribute__((ext_vector_type(4)))  unsigned int u32x4;

#define NSEQ   24
#define LSEQ   512
#define DMODEL 96
#define DINNER 192
#define DSTATE 64
#define NHEADS 8
#define HD     24
#define CONVD  320
#define PROJ   520
#define PROJP  528
#define NTOK   (NSEQ*LSEQ)
#define EPSV   1e-5f

__device__ __forceinline__ float siluf(float x){ return x / (1.f + __expf(-x)); }

// ---------------------------------------------------------------------------
// WMMA fragment loaders (16x16x32 f16, wave32 layouts per CDNA5 ISA 7.12.2)
// A fragment: lane(0..15)=row M, halves e<8 -> K=e+8g, e>=8 -> K=16+(e-8)+8g
//   => two contiguous 8-half (16B) runs: [k0+8g, +8) and [k0+16+8g, +8)
// B fragment: lane&15 = col N (weight row), same K order as A
//   => two contiguous 8-half (16B) runs: [k0+8g, +8) and [k0+16+8g, +8)
// ---------------------------------------------------------------------------
__device__ __forceinline__ v16h load_a_frag(const _Float16* __restrict__ A, int lda,
                                            int row0, int k0){
  int lane = threadIdx.x & 31;
  int g = lane >> 4;
  const _Float16* p = A + (size_t)(row0 + (lane & 15)) * lda + k0 + 8*g;
  union { u32x4 q[2]; v16h v; } u;
  u.q[0] = *reinterpret_cast<const u32x4*>(p);
  u.q[1] = *reinterpret_cast<const u32x4*>(p + 16);
  return u.v;
}
__device__ __forceinline__ v16h load_b_frag(const _Float16* __restrict__ W, int ldw,
                                            int n0, int k0){
  int lane = threadIdx.x & 31;
  int g = lane >> 4;
  const _Float16* p = W + (size_t)(n0 + (lane & 15)) * ldw + k0 + 8*g;
  union { u32x4 q[2]; v16h v; } u;
  u.q[0] = *reinterpret_cast<const u32x4*>(p);
  u.q[1] = *reinterpret_cast<const u32x4*>(p + 16);
  return u.v;
}

__device__ __forceinline__ v8f wmma_step(v16h a, v16h b, v8f c){
  return __builtin_amdgcn_wmma_f32_16x16x32_f16(
      /*neg_a=*/false, a, /*neg_b=*/false, b,
      /*c_mod=*/(short)0, c, /*reuse_a=*/false, /*reuse_b=*/false);
}

// Epilogue modes
#define EP_DSRED  0   // store transposed into res (B,96,512)
#define EP_PLAIN  1   // f32 store, Nreal bound check
#define EP_ADDRES 2   // f32 in-place add (residual += h)
#define EP_GELU16 3   // f16 store of gelu(acc + bias)
#define EP_BIAS   4   // f32 store of acc + bias

// One wave computes a 2x3 block of 16x16 tiles: C = A[M,K] * W[N,K]^T.
// 6 WMMAs per 5 fragments per k-step (A reused x3, B reused x2).
__global__ __launch_bounds__(32)
void gemm_wmma(const _Float16* __restrict__ A, int lda,
               const _Float16* __restrict__ W, int ldw,
               int K, int Nreal, int mode,
               float* __restrict__ outf, int ldo,
               _Float16* __restrict__ outh,
               const float* __restrict__ bias)
{
  int mt = blockIdx.x * 2;           // first of 2 M-tiles
  int nt = blockIdx.y * 3;           // first of 3 N-tiles
  int lane = threadIdx.x & 31;
  v8f acc[2][3] = {};
  for (int k0 = 0; k0 < K; k0 += 32) {
    v16h a0 = load_a_frag(A, lda, (mt+0)*16, k0);
    v16h a1 = load_a_frag(A, lda, (mt+1)*16, k0);
    v16h b0 = load_b_frag(W, ldw, (nt+0)*16, k0);
    v16h b1 = load_b_frag(W, ldw, (nt+1)*16, k0);
    v16h b2 = load_b_frag(W, ldw, (nt+2)*16, k0);
    acc[0][0] = wmma_step(a0, b0, acc[0][0]);
    acc[0][1] = wmma_step(a0, b1, acc[0][1]);
    acc[0][2] = wmma_step(a0, b2, acc[0][2]);
    acc[1][0] = wmma_step(a1, b0, acc[1][0]);
    acc[1][1] = wmma_step(a1, b1, acc[1][1]);
    acc[1][2] = wmma_step(a1, b2, acc[1][2]);
  }
  #pragma unroll
  for (int mi = 0; mi < 2; ++mi) {
    #pragma unroll
    for (int ni = 0; ni < 3; ++ni) {
      int col = (nt + ni)*16 + (lane & 15);
      if (col >= Nreal) continue;
      int rbase = (mt + mi)*16 + 8*(lane >> 4);
      #pragma unroll
      for (int r = 0; r < 8; ++r) {
        int row = rbase + r;
        float v = acc[mi][ni][r];
        if (mode == EP_DSRED) {
          int b2i = row >> 9, t = row & 511;                // row = b*512 + token
          outf[((size_t)(b2i*DMODEL + col))*512 + t] = v;   // res[b][col][t]
        } else if (mode == EP_PLAIN) {
          outf[(size_t)row*ldo + col] = v;
        } else if (mode == EP_ADDRES) {
          size_t idx = (size_t)row*ldo + col;
          outf[idx] = v + outf[idx];
        } else if (mode == EP_GELU16) {
          float xg = v + bias[col];
          outh[(size_t)row*ldo + col] =
              (_Float16)(0.5f*xg*(1.0f + erff(xg*0.70710678118f)));
        } else { // EP_BIAS
          outf[(size_t)row*ldo + col] = v + bias[col];
        }
      }
    }
  }
}

// f32 -> f16 weight conversion with zero padding (for N=520 -> 528)
__global__ void convert_w(const float* __restrict__ src, _Float16* __restrict__ dst,
                          int n_src, int n_dst){
  int i = blockIdx.x*blockDim.x + threadIdx.x;
  if (i < n_dst) dst[i] = (i < n_src) ? (_Float16)src[i] : (_Float16)0.f;
}

// downsample gather index: feature f = c*8 + di*4 + dj*2 + dk at (2i+di,2j+dj,2k+dk)
__device__ __forceinline__ size_t ds_idx(int b, int i, int j, int k, int f){
  int c = f >> 3, di = (f >> 2) & 1, dj = (f >> 1) & 1, dk = f & 1;
  return ((((size_t)b*48 + c)*16 + (2*i + di))*16 + (2*j + dj))*16 + (2*k + dk);
}

// Downsample + LayerNorm(384) -> f16 tokens for ds_red GEMM
__global__ void ds_ln_pack(const float* __restrict__ x, const float* __restrict__ w,
                           const float* __restrict__ b, _Float16* __restrict__ T16){
  int tok = blockIdx.x*blockDim.x + threadIdx.x;
  if (tok >= 1024) return;
  int bb = tok >> 9, p = tok & 511;
  int i = p >> 6, j = (p >> 3) & 7, k = p & 7;
  float m = 0.f;
  for (int f = 0; f < 384; ++f) m += x[ds_idx(bb,i,j,k,f)];
  m *= (1.f/384.f);
  float var = 0.f;
  for (int f = 0; f < 384; ++f){ float d = x[ds_idx(bb,i,j,k,f)] - m; var += d*d; }
  float rs = rsqrtf(var*(1.f/384.f) + EPSV);
  for (int f = 0; f < 384; ++f)
    T16[(size_t)tok*384 + f] = (_Float16)((x[ds_idx(bb,i,j,k,f)] - m)*rs*w[f] + b[f]);
}

// scan-order map: l -> spatial position (forward), per scan q in [0,12)
__device__ __forceinline__ int fwd_pos(int q, int l){
  int v = q & 3;
  if (v >= 2) l = 511 - l;
  int a = l >> 6, c = (l >> 3) & 7, e = l & 7;
  int o = v & 1, g = q >> 2;
  int i, j, k;
  if (g == 0){ if(!o){ i=a; j=c; k=e; } else { i=a; k=c; j=e; } }
  else if (g == 1){ if(!o){ j=a; k=c; i=e; } else { j=a; i=c; k=e; } }
  else { if(!o){ k=a; i=c; j=e; } else { k=a; j=c; i=e; } }
  return i*64 + j*8 + k;
}
__device__ __forceinline__ int inv_l(int q, int i, int j, int k){
  int g = q >> 2, v = q & 3, o = v & 1;
  int l;
  if (g == 0) l = o ? (i*64 + k*8 + j) : (i*64 + j*8 + k);
  else if (g == 1) l = o ? (j*64 + i*8 + k) : (j*64 + k*8 + i);
  else            l = o ? (k*64 + j*8 + i) : (k*64 + i*8 + j);
  return (v >= 2) ? (511 - l) : l;
}

__global__ void build_hs(const float* __restrict__ res, float* __restrict__ hs){
  int idx = blockIdx.x*blockDim.x + threadIdx.x;
  if (idx >= NTOK*DMODEL) return;
  int d = idx % DMODEL;
  int sl = idx / DMODEL;
  int l = sl % LSEQ, s = sl / LSEQ;
  int b = s / 12, q = s % 12;
  hs[idx] = res[((size_t)b*DMODEL + d)*512 + fwd_pos(q, l)];
}

__global__ void add_or_copy(float* __restrict__ dst, const float* __restrict__ src,
                            int add, int n){
  int i = blockIdx.x*blockDim.x + threadIdx.x;
  if (i < n) dst[i] = add ? (dst[i] + src[i]) : src[i];
}

// per-token LayerNorm over `width`; writes f16 (outh) or f32 (outf)
__global__ void ln_token(const float* __restrict__ in, const float* __restrict__ w,
                         const float* __restrict__ b, int ntok, int width,
                         _Float16* __restrict__ outh, float* __restrict__ outf){
  int t = blockIdx.x*blockDim.x + threadIdx.x;
  if (t >= ntok) return;
  const float* p = in + (size_t)t*width;
  float m = 0.f;
  for (int f = 0; f < width; ++f) m += p[f];
  m /= width;
  float v = 0.f;
  for (int f = 0; f < width; ++f){ float d = p[f] - m; v += d*d; }
  float rs = rsqrtf(v/width + EPSV);
  for (int f = 0; f < width; ++f){
    float y = (p[f] - m)*rs*w[f] + b[f];
    if (outh) outh[(size_t)t*width + f] = (_Float16)y;
    else      outf[(size_t)t*width + f] = y;
  }
}

// depthwise causal conv (D_CONV=4) + bias + SiLU over the 320 xBC channels
__global__ void conv_silu(const float* __restrict__ zx, const float* __restrict__ cw,
                          const float* __restrict__ cb, float* __restrict__ xc){
  int idx = blockIdx.x*blockDim.x + threadIdx.x;
  if (idx >= NTOK*CONVD) return;
  int ch = idx % CONVD;
  int tok = idx / CONVD;
  int l = tok % LSEQ, s = tok / LSEQ;
  float acc = cb[ch];
  #pragma unroll
  for (int k = 0; k < 4; ++k){
    int lp = l + k - 3;
    if (lp >= 0)
      acc += zx[((size_t)s*LSEQ + lp)*PROJ + DINNER + ch] * cw[ch*4 + k];
  }
  xc[idx] = siluf(acc);
}

// dt = softplus(raw + dt_bias), dA = exp(-exp(A_log)*dt)
__global__ void dt_kernel(const float* __restrict__ zx, const float* __restrict__ dtb,
                          const float* __restrict__ alog,
                          float* __restrict__ dtbuf, float* __restrict__ dabuf){
  int idx = blockIdx.x*blockDim.x + threadIdx.x;
  if (idx >= NTOK*NHEADS) return;
  int h = idx % NHEADS;
  float raw = zx[(size_t)(idx/NHEADS)*PROJ + 512 + h] + dtb[h];
  float sp = (raw > 20.f) ? raw : log1pf(__expf(raw));
  dtbuf[idx] = sp;
  dabuf[idx] = __expf(-__expf(alog[h]) * sp);
}

// Sequential SSM scan: 1 block / sequence, 192 threads = (head, headdim) pairs,
// 64-wide state in registers; B_t/C_t staged in LDS; fused SiLU gate + RMSNorm.
__global__ __launch_bounds__(192)
void ssm_scan(const float* __restrict__ zx, const float* __restrict__ xc,
              const float* __restrict__ dtbuf, const float* __restrict__ dabuf,
              const float* __restrict__ Dv, const float* __restrict__ nw,
              _Float16* __restrict__ yout)
{
  int s = blockIdx.x;
  int tid = threadIdx.x;               // 0..191 = h*24 + p
  int h = tid / HD;
  __shared__ float Bs[DSTATE], Cs[DSTATE], red[DINNER];
  float hst[DSTATE];
  #pragma unroll
  for (int n = 0; n < DSTATE; ++n) hst[n] = 0.f;
  float dcoef = Dv[h];
  float nwv = nw[tid];
  for (int t = 0; t < LSEQ; ++t){
    size_t tb = (size_t)(s*LSEQ + t);
    if (tid < 64)       Bs[tid]    = xc[tb*CONVD + DINNER + tid];
    else if (tid < 128) Cs[tid-64] = xc[tb*CONVD + DINNER + DSTATE + (tid-64)];
    __syncthreads();
    float dtv = dtbuf[tb*NHEADS + h];
    float dav = dabuf[tb*NHEADS + h];
    float xv  = xc[tb*CONVD + tid];
    float dtx = dtv * xv;
    float acc = 0.f;
    #pragma unroll
    for (int n = 0; n < DSTATE; ++n){
      hst[n] = dav*hst[n] + dtx*Bs[n];
      acc += hst[n]*Cs[n];
    }
    float y = acc + dcoef*xv;
    float zv = zx[tb*PROJ + tid];
    y *= siluf(zv);
    red[tid] = y*y;
    __syncthreads();
    if (tid < 96) red[tid] += red[tid+96];  __syncthreads();
    if (tid < 48) red[tid] += red[tid+48];  __syncthreads();
    if (tid < 24) red[tid] += red[tid+24];  __syncthreads();
    if (tid < 12) red[tid] += red[tid+12];  __syncthreads();
    if (tid <  6) red[tid] += red[tid+ 6];  __syncthreads();
    if (tid == 0) red[0] = red[0]+red[1]+red[2]+red[3]+red[4]+red[5];
    __syncthreads();
    float ms = red[0] * (1.f/(float)DINNER);
    yout[tb*DINNER + tid] = (_Float16)(y * rsqrtf(ms + EPSV) * nwv);
    __syncthreads();   // protect Bs/Cs/red before next timestep
  }
}

__global__ void merge_out(const float* __restrict__ hsf, float* __restrict__ out){
  int idx = blockIdx.x*blockDim.x + threadIdx.x;
  if (idx >= 2*DMODEL*512) return;
  int pos = idx % 512;
  int d = (idx/512) % DMODEL;
  int b = idx / (512*DMODEL);
  int i = pos >> 6, j = (pos >> 3) & 7, k = pos & 7;
  float acc = 0.f;
  #pragma unroll
  for (int q = 0; q < 12; ++q){
    int l = inv_l(q, i, j, k);
    acc += hsf[(((size_t)(b*12 + q))*LSEQ + l)*DMODEL + d];
  }
  out[idx] = acc * (1.f/12.f);
}

extern "C" void kernel_launch(void* const* d_in, const int* in_sizes, int n_in,
                              void* d_out, int out_size, void* d_ws, size_t ws_size,
                              hipStream_t stream)
{
  (void)in_sizes; (void)n_in; (void)out_size; (void)ws_size;
  const float* x     = (const float*)d_in[0];
  const float* ds_w  = (const float*)d_in[1];
  const float* ds_b  = (const float*)d_in[2];
  const float* dsred = (const float*)d_in[3];
  const float* n1w   = (const float*)d_in[4];
  const float* n1b   = (const float*)d_in[5];
  const float* inw   = (const float*)d_in[6];
  const float* cw    = (const float*)d_in[7];
  const float* cb    = (const float*)d_in[8];
  const float* dtb   = (const float*)d_in[9];
  const float* alog  = (const float*)d_in[10];
  const float* Dv    = (const float*)d_in[11];
  const float* snw   = (const float*)d_in[12];
  const float* outw  = (const float*)d_in[13];
  const float* n2w   = (const float*)d_in[14];
  const float* n2b   = (const float*)d_in[15];
  const float* f1w   = (const float*)d_in[16];
  const float* f1b   = (const float*)d_in[17];
  const float* f2w   = (const float*)d_in[18];
  const float* f2b   = (const float*)d_in[19];
  const float* nfw   = (const float*)d_in[20];
  const float* nfb   = (const float*)d_in[21];
  float* out = (float*)d_out;

  // workspace carve-up (256B aligned); total ~67 MB
  char* ws = (char*)d_ws;
  size_t off = 0;
  auto carve = [&](size_t bytes)->char*{
    char* p = ws + off; off += (bytes + 255) & ~(size_t)255; return p;
  };
  _Float16* T16    = (_Float16*)carve((size_t)1024*384*2);
  _Float16* wds16  = (_Float16*)carve((size_t)96*384*2);
  _Float16* win16  = (_Float16*)carve((size_t)2*PROJP*96*2);
  _Float16* wout16 = (_Float16*)carve((size_t)2*96*192*2);
  _Float16* wf116  = (_Float16*)carve((size_t)2*192*96*2);
  _Float16* wf216  = (_Float16*)carve((size_t)2*96*192*2);
  float* res   = (float*)carve((size_t)2*96*512*4);
  float* hs    = (float*)carve((size_t)NTOK*96*4);
  float* resid = (float*)carve((size_t)NTOK*96*4);
  float* zx    = (float*)carve((size_t)NTOK*PROJ*4);
  float* xc    = (float*)carve((size_t)NTOK*CONVD*4);
  float* dtbuf = (float*)carve((size_t)NTOK*8*4);
  float* dabuf = (float*)carve((size_t)NTOK*8*4);
  _Float16* X16 = (_Float16*)carve((size_t)NTOK*192*2);
  _Float16* Y16 = (_Float16*)carve((size_t)NTOK*192*2);
  float* hsf   = (float*)carve((size_t)NTOK*96*4);

  auto cvt = [&](const float* s, _Float16* d, int nsrc, int ndst){
    convert_w<<<(ndst+255)/256, 256, 0, stream>>>(s, d, nsrc, ndst);
  };
  cvt(dsred, wds16, 96*384, 96*384);
  for (int l = 0; l < 2; ++l){
    cvt(inw  + l*520*96,  win16  + l*PROJP*96, 520*96, PROJP*96);
    cvt(outw + l*96*192,  wout16 + l*96*192,   96*192, 96*192);
    cvt(f1w  + l*192*96,  wf116  + l*192*96,   192*96, 192*96);
    cvt(f2w  + l*96*192,  wf216  + l*96*192,   96*192, 96*192);
  }

  // downsample: LN(384) -> GEMM(1024x384x96) -> res (B,96,512) -> 12 scan orders
  ds_ln_pack<<<4, 256, 0, stream>>>(x, ds_w, ds_b, T16);
  {
    dim3 g(1024/16/2, 96/16/3);  // (32, 2)
    gemm_wmma<<<g, 32, 0, stream>>>(T16, 384, wds16, 384, 384, 96, EP_DSRED,
                                    res, 0, nullptr, nullptr);
  }
  build_hs<<<(NTOK*96+255)/256, 256, 0, stream>>>(res, hs);

  for (int l = 0; l < 2; ++l){
    add_or_copy<<<(NTOK*96+255)/256, 256, 0, stream>>>(resid, hs, l > 0 ? 1 : 0, NTOK*96);
    // norm1 -> in_proj (M=12288, K=96, N=520 pad 528)
    ln_token<<<(NTOK+255)/256, 256, 0, stream>>>(resid, n1w + l*96, n1b + l*96,
                                                 NTOK, 96, X16, nullptr);
    { dim3 g(NTOK/16/2, PROJP/16/3);   // (384, 11)
      gemm_wmma<<<g, 32, 0, stream>>>(X16, 96, win16 + l*PROJP*96, 96,
                                      96, PROJ, EP_PLAIN, zx, PROJ, nullptr, nullptr); }
    conv_silu<<<(NTOK*CONVD+255)/256, 256, 0, stream>>>(zx, cw + l*CONVD*4, cb + l*CONVD, xc);
    dt_kernel<<<(NTOK*8+255)/256, 256, 0, stream>>>(zx, dtb + l*8, alog + l*8, dtbuf, dabuf);
    ssm_scan<<<NSEQ, 192, 0, stream>>>(zx, xc, dtbuf, dabuf, Dv + l*8, snw + l*192, X16);
    // out_proj (M=12288, K=192, N=96) fused residual add
    { dim3 g(NTOK/16/2, 96/16/3);      // (384, 2)
      gemm_wmma<<<g, 32, 0, stream>>>(X16, 192, wout16 + l*96*192, 192,
                                      192, 96, EP_ADDRES, resid, 96, nullptr, nullptr); }
    // norm2 -> fc1(+gelu) -> fc2(+bias) -> hs
    ln_token<<<(NTOK+255)/256, 256, 0, stream>>>(resid, n2w + l*96, n2b + l*96,
                                                 NTOK, 96, X16, nullptr);
    { dim3 g(NTOK/16/2, 192/16/3);     // (384, 4)
      gemm_wmma<<<g, 32, 0, stream>>>(X16, 96, wf116 + l*192*96, 96,
                                      96, 192, EP_GELU16, nullptr, 192, Y16, f1b + l*192); }
    { dim3 g(NTOK/16/2, 96/16/3);      // (384, 2)
      gemm_wmma<<<g, 32, 0, stream>>>(Y16, 192, wf216 + l*96*192, 192,
                                      192, 96, EP_BIAS, hs, 96, nullptr, f2b + l*96); }
  }

  // final: LN(hs + residual) then merge the 12 scan directions
  add_or_copy<<<(NTOK*96+255)/256, 256, 0, stream>>>(hs, resid, 1, NTOK*96);
  ln_token<<<(NTOK+255)/256, 256, 0, stream>>>(hs, nfw, nfb, NTOK, 96, nullptr, hsf);
  merge_out<<<(2*96*512+255)/256, 256, 0, stream>>>(hsf, out);
}
